// LearnableOde_18957985644608
// MI455X (gfx1250) — compile-verified, matrix operand order checked by reading, not run on hardware
//
#include <hip/hip_runtime.h>

typedef _Float16 v16h __attribute__((ext_vector_type(16)));
typedef _Float16 v8h  __attribute__((ext_vector_type(8)));
typedef float    v8f  __attribute__((ext_vector_type(8)));
typedef float    v4f  __attribute__((ext_vector_type(4)));

#define WAIT_DS() asm volatile("s_wait_dscnt 0" ::: "memory")

constexpr int Dx = 64, Dz = 128, Dy = 10, NSTEPS = 40;
constexpr float DT = 1.0f / 40.0f;   // (T1 - T0) / N_STEPS
constexpr int ROWS_PB = 128;         // rows of B handled per block (8 waves x 16)
constexpr int THREADS = 256;

// Hardware transcendental tanh (CDNA5 TRANS32 op). ISA 7.4: one independent
// op / V_NOP required after a TRANS op before its result is consumed; the
// compiler cannot know that for inline asm, so bake the V_NOP in.
static __device__ __forceinline__ float htanh(float x) {
  float r;
  asm("v_tanh_f32 %0, %1\n\tv_nop" : "=v"(r) : "v"(x));
  return r;
}

// ---- WMMA: D(16x16,f32) = A(16x32,f16) x B(32x16,f16) + C ----
static __device__ __forceinline__ v8f wmma_f16(v16h a, v16h b, v8f c) {
  return __builtin_amdgcn_wmma_f32_16x16x32_f16(false, a, false, b, (short)0, c,
                                                false, false);
}

// Weight matrices pre-swizzled into B-fragment layout, f16.
// P: K=64 -> 2 K-chunks; W1/W2: K=128 -> 4 K-chunks; 8 N-tiles each; 512 halves/frag.
struct WFrags {
  _Float16 Pf [2 * 8 * 512];   // 16 KB
  _Float16 W1f[4 * 8 * 512];   // 32 KB
  _Float16 W2f[4 * 8 * 512];   // 32 KB (stores dt*W2)
};
union SmemU {
  WFrags w;                    // live during projection + ODE loop
  float ztile[ROWS_PB * Dz];   // 64 KB, reused for the readout stage
};

// K-offset mapping of the 16-bit 16x32 A / 32x16 B fragment (ISA 7.12.2):
// lane<16 holds K {0..7,16..23}; lane>=16 holds K {8..15,24..31}; j indexes halves.
__device__ __forceinline__ int k_of(int lane, int j) {
  return ((lane >= 16) ? 8 : 0) + ((j >= 8) ? 16 : 0) + (j & 7);
}

// Convert a K-major (K x 128, row-major, f32) weight matrix into f16 B-fragments,
// with an optional constant scale folded in (dt for W2).
__device__ __forceinline__ void fill_b_frags(_Float16* dst, const float* src,
                                             int nkc, int tid, float scale) {
  const int total = nkc * 8 * 512;
  for (int e = tid; e < total; e += THREADS) {
    int frag = e >> 9;                 // kc*8 + nt
    int kc = frag >> 3, nt = frag & 7;
    int le = e & 511;
    int lane = le >> 4, j = le & 15;
    int k = kc * 32 + k_of(lane, j);
    int n = nt * 16 + (lane & 15);
    dst[e] = (_Float16)(scale * src[k * Dz + n]);
  }
}

// B fragment: lane-contiguous 32B chunk -> one v16h load (2x ds_load_b128).
__device__ __forceinline__ v16h load_b_frag(const _Float16* base, int kc, int nt,
                                            int lane) {
  return *(const v16h*)(base + ((kc * 8 + nt) << 9) + lane * 16);
}

// A fragment from a row-major 16x128 f16 staging tile in LDS.
__device__ __forceinline__ v16h load_a_frag_lds(const _Float16* st, int kc,
                                                int lane) {
  int m = lane & 15;
  int ko = kc * 32 + ((lane >= 16) ? 8 : 0);
  const _Float16* p = st + m * Dz + ko;
  v8h lo = *(const v8h*)p;          // K group {+0..+7}
  v8h hi = *(const v8h*)(p + 16);   // K group {+16..+23}
  return __builtin_shufflevector(lo, hi, 0, 1, 2, 3, 4, 5, 6, 7, 8, 9, 10, 11,
                                 12, 13, 14, 15);
}

// A fragment straight from global x (row-major B x 64, f32), converted to f16.
__device__ __forceinline__ v16h load_x_frag(const float* xrow, int kc, int lane) {
  int ko = kc * 32 + ((lane >= 16) ? 8 : 0);
  v4f g0 = *(const v4f*)(xrow + ko);
  v4f g1 = *(const v4f*)(xrow + ko + 4);
  v4f g2 = *(const v4f*)(xrow + ko + 16);
  v4f g3 = *(const v4f*)(xrow + ko + 20);
  v16h a;
  for (int i = 0; i < 4; ++i) {
    a[i]      = (_Float16)g0[i];
    a[4 + i]  = (_Float16)g1[i];
    a[8 + i]  = (_Float16)g2[i];
    a[12 + i] = (_Float16)g3[i];
  }
  return a;
}

__global__ __launch_bounds__(THREADS) void node_kernel(
    const float* __restrict__ x, const float* __restrict__ P,
    const float* __restrict__ W1, const float* __restrict__ b1,
    const float* __restrict__ W2, const float* __restrict__ b2,
    const float* __restrict__ Qw, const float* __restrict__ Qb,
    float* __restrict__ out) {
  __shared__ SmemU smem;                    // 80 KB (union)
  __shared__ _Float16 stageA[8][16 * Dz];   // 32 KB: per-wave A staging
  __shared__ float sQw[Dz * Dy];            // 5 KB

  const int tid = threadIdx.x;
  const int lane = tid & 31;
  const int wv = tid >> 5;
  const int col = lane & 15;                // N column this lane owns in C frags
  const int mbase = (lane < 16) ? 0 : 8;    // M row base in C frags

  // ---- one-time weight staging (f32 -> f16 B-fragments in LDS) ----
  fill_b_frags(smem.w.Pf,  P,  2, tid, 1.0f);
  fill_b_frags(smem.w.W1f, W1, 4, tid, 1.0f);
  fill_b_frags(smem.w.W2f, W2, 4, tid, DT);   // fold dt into W2
  for (int e = tid; e < Dz * Dy; e += THREADS) sQw[e] = Qw[e];
  __syncthreads();

  // per-lane bias values for the C-fragment columns this lane owns
  float b1v[8], b2v[8];
  for (int nt = 0; nt < 8; ++nt) {
    b1v[nt] = b1[nt * 16 + col];
    b2v[nt] = DT * b2[nt * 16 + col];   // fold dt into the b2 add
  }

  const int rowbase = blockIdx.x * ROWS_PB + wv * 16;

  // ---- z0 = x @ P (f32 accumulators, resident for all 40 steps) ----
  v8f z[8];
  for (int nt = 0; nt < 8; ++nt)
    for (int r = 0; r < 8; ++r) z[nt][r] = 0.0f;
  {
    const float* xrow = x + (size_t)(rowbase + col) * Dx;
    for (int kc = 0; kc < 2; ++kc) {
      v16h a = load_x_frag(xrow, kc, lane);
      for (int nt = 0; nt < 8; ++nt) {
        v16h b = load_b_frag(smem.w.Pf, kc, nt, lane);
        z[nt] = wmma_f16(a, b, z[nt]);
      }
    }
  }

  _Float16* myst = stageA[wv];   // this wave's private 16x128 staging tile

  // ---- 40 forward-Euler steps, fully on-chip ----
  for (int step = 0; step < NSTEPS; ++step) {
    // stage z as f16 A-operand (C-layout -> row-major via LDS)
    for (int nt = 0; nt < 8; ++nt)
      for (int r = 0; r < 8; ++r)
        myst[(mbase + r) * Dz + nt * 16 + col] = (_Float16)z[nt][r];
    WAIT_DS();

    // h = z @ W1 + b1
    v8f h[8];
    for (int nt = 0; nt < 8; ++nt)
      for (int r = 0; r < 8; ++r) h[nt][r] = b1v[nt];
    for (int kc = 0; kc < 4; ++kc) {
      v16h a = load_a_frag_lds(myst, kc, lane);
      for (int nt = 0; nt < 8; ++nt) {
        v16h b = load_b_frag(smem.w.W1f, kc, nt, lane);
        h[nt] = wmma_f16(a, b, h[nt]);
      }
    }

    // stage tanh(h) as f16 A-operand (dt already folded into W2 fragments)
    for (int nt = 0; nt < 8; ++nt)
      for (int r = 0; r < 8; ++r)
        myst[(mbase + r) * Dz + nt * 16 + col] = (_Float16)htanh(h[nt][r]);
    WAIT_DS();

    // z += tanh(h) @ (dt*W2), then z += dt*b2
    for (int kc = 0; kc < 4; ++kc) {
      v16h a = load_a_frag_lds(myst, kc, lane);
      for (int nt = 0; nt < 8; ++nt) {
        v16h b = load_b_frag(smem.w.W2f, kc, nt, lane);
        z[nt] = wmma_f16(a, b, z[nt]);
      }
    }
    for (int nt = 0; nt < 8; ++nt)
      for (int r = 0; r < 8; ++r) z[nt][r] += b2v[nt];
  }

  // ---- readout: y = tanh(zT) @ Qw + Qb ----
  __syncthreads();   // everyone done with weight frags before union reuse
  for (int nt = 0; nt < 8; ++nt)
    for (int r = 0; r < 8; ++r)
      smem.ztile[(wv * 16 + mbase + r) * Dz + nt * 16 + col] =
          htanh(z[nt][r]);
  __syncthreads();

  {
    int row = tid & 127;          // 2 threads per row
    int ob = (tid >> 7) * 5;      // 5 outputs each (Dy = 10)
    const float* zr = smem.ztile + row * Dz;
    float acc[5];
    for (int o = 0; o < 5; ++o) acc[o] = Qb[ob + o];
    for (int k = 0; k < Dz; ++k) {
      float zv = zr[k];
      for (int o = 0; o < 5; ++o) acc[o] += zv * sQw[k * Dy + ob + o];
    }
    float* orow = out + (size_t)(blockIdx.x * ROWS_PB + row) * Dy + ob;
    for (int o = 0; o < 5; ++o) orow[o] = acc[o];
  }
}

extern "C" void kernel_launch(void* const* d_in, const int* in_sizes, int n_in,
                              void* d_out, int out_size, void* d_ws,
                              size_t ws_size, hipStream_t stream) {
  const float* x  = (const float*)d_in[0];
  const float* P  = (const float*)d_in[1];
  const float* W1 = (const float*)d_in[2];
  const float* b1 = (const float*)d_in[3];
  const float* W2 = (const float*)d_in[4];
  const float* b2 = (const float*)d_in[5];
  const float* Qw = (const float*)d_in[6];
  const float* Qb = (const float*)d_in[7];
  float* out = (float*)d_out;

  const int B = in_sizes[0] / Dx;         // 65536
  const int blocks = B / ROWS_PB;         // 512
  hipLaunchKernelGGL(node_kernel, dim3(blocks), dim3(THREADS), 0, stream, x, P,
                     W1, b1, W2, b2, Qw, Qb, out);
}
